// KANLayer_78005196030433
// MI455X (gfx1250) — matrix operand order repaired and checked by reading, not compiled, not run on hardware
//
#include <hip/hip_runtime.h>
#include <cstdint>

// ---------------------------------------------------------------------------
// KAN layer as one f16 WMMA GEMM:
//   out(512x512) = Aaug(512x6144) @ Bt^T   (Bt stored as 512x6144, row = out col)
//   Aaug[b][i*12+k] = spline basis_k(x[b,i]) (k<11), silu(x[b,i]) (k==11)
//   Bt  [o][i*12+k] = mask*scale_sp*coef[i,o,k] (k<11), mask*scale_base (k==11)
// ---------------------------------------------------------------------------

typedef __attribute__((ext_vector_type(16))) _Float16 v16h;
typedef __attribute__((ext_vector_type(8)))  float    v8f;

#define BATCH 512
#define IN_DIM 512
#define OUT_DIM 512
#define KAUG 6144            // 512 * 12
#define SPLITK 4
#define K_PER_SPLIT (KAUG / SPLITK)   // 1536
#define BM 64
#define BN 64
#define BK 64
#define LDT 72               // padded LDS row stride in halves (144 B, 16B-aligned)
#define TILE (BM * LDT)      // halves per LDS tile

union Frag { v16h v; uint64_t q[4]; };
union Pack12 { _Float16 h[12]; uint64_t q[3]; };

// ---------------------------- prep kernels ---------------------------------

__global__ __launch_bounds__(256) void kan_zero_out(float* __restrict__ out) {
    int idx = blockIdx.x * 256 + threadIdx.x;   // exactly 512*512 threads
    out[idx] = 0.0f;
}

__global__ __launch_bounds__(256) void kan_prep_a(const float* __restrict__ x,
                                                  const float* __restrict__ grid,
                                                  _Float16* __restrict__ Aaug) {
    int idx = blockIdx.x * 256 + threadIdx.x;   // 0 .. 512*512-1
    int b = idx >> 9, i = idx & 511;
    float xv = x[b * IN_DIM + i];

    float g[15];
#pragma unroll
    for (int t = 0; t < 15; ++t) g[t] = grid[i * 15 + t];

    float Bv[14];
#pragma unroll
    for (int t = 0; t < 14; ++t) Bv[t] = (xv >= g[t] && xv < g[t + 1]) ? 1.0f : 0.0f;
#pragma unroll
    for (int j = 1; j <= 3; ++j) {
#pragma unroll
        for (int t = 0; t < 14 - 1; ++t) {      // only first 14-j entries meaningful
            if (t < 14 - j) {
                float left  = (xv - g[t])       / (g[t + j]     - g[t]);
                float right = (g[t + j + 1] - xv) / (g[t + j + 1] - g[t + 1]);
                Bv[t] = left * Bv[t] + right * Bv[t + 1];
            }
        }
    }
    float sig = 1.0f / (1.0f + __expf(-xv));

    Pack12 u;
#pragma unroll
    for (int t = 0; t < 11; ++t) u.h[t] = (_Float16)Bv[t];
    u.h[11] = (_Float16)(xv * sig);

    uint64_t* dst = (uint64_t*)(Aaug + (size_t)b * KAUG + i * 12);
    dst[0] = u.q[0]; dst[1] = u.q[1]; dst[2] = u.q[2];
}

__global__ __launch_bounds__(256) void kan_prep_b(const float* __restrict__ coef,
                                                  const float* __restrict__ sb,
                                                  const float* __restrict__ ss,
                                                  const float* __restrict__ mask,
                                                  _Float16* __restrict__ Bt) {
    int idx = blockIdx.x * 256 + threadIdx.x;   // 0 .. 512*512-1
    int i = idx >> 9, o = idx & 511;
    int io = i * OUT_DIM + o;
    float m   = mask[io];
    float wsp = m * ss[io];
    float wb  = m * sb[io];
    const float* c = coef + (size_t)io * 11;

    Pack12 u;
#pragma unroll
    for (int t = 0; t < 11; ++t) u.h[t] = (_Float16)(wsp * c[t]);
    u.h[11] = (_Float16)wb;

    uint64_t* dst = (uint64_t*)(Bt + (size_t)o * KAUG + i * 12);
    dst[0] = u.q[0]; dst[1] = u.q[1]; dst[2] = u.q[2];
}

// ---------------------------- WMMA GEMM ------------------------------------
// grid = (8, 8, SPLITK), block = 256 threads = 8 waves.
// Wave (wm,wn): wm = wave&3 -> 16 rows of M, wn = wave>>2 -> 32 cols of N.

__global__ __launch_bounds__(256) void kan_wmma_gemm(const _Float16* __restrict__ A,
                                                     const _Float16* __restrict__ Bt,
                                                     float* __restrict__ out) {
    __shared__ _Float16 lds[2 * 2 * TILE];      // double buffer x (Atile, Btile)

    const int tid   = threadIdx.x;
    const int wave  = tid >> 5;
    const int lane  = tid & 31;
    const int laneLo = lane & 15;
    const int laneHi = lane >> 4;
    const int wm = wave & 3;
    const int wn = wave >> 2;

    const int m0    = blockIdx.x * BM;
    const int n0    = blockIdx.y * BN;
    const int kBase = blockIdx.z * K_PER_SPLIT;
    const int nStages = K_PER_SPLIT / BK;       // 24

    const uint32_t ldsBase = (uint32_t)(uintptr_t)(&lds[0]);

    const uint64_t gABase = (uint64_t)(uintptr_t)(A  + (size_t)m0 * KAUG + kBase);
    const uint64_t gBBase = (uint64_t)(uintptr_t)(Bt + (size_t)n0 * KAUG + kBase);

    // async-stage one 64x64 f16 tile of A and of Bt into LDS buffer `buf`
    auto stage_load = [&](int buf, int s) {
        const uint32_t aLds = ldsBase + (uint32_t)(buf * 2 * TILE) * 2u;
        const uint32_t bLds = aLds + (uint32_t)TILE * 2u;
        const uint64_t kOff = (uint64_t)(s * BK) * 2u;      // byte offset along K
#pragma unroll
        for (int t = 0; t < 2; ++t) {
            int chunk = tid + t * 256;          // 0..511 : 64 rows x 8 chunks(16B)
            int row = chunk >> 3;
            int col = (chunk & 7) * 8;          // halves
            uint64_t ga = gABase + kOff + ((uint64_t)row * KAUG + col) * 2u;
            uint32_t la = aLds + (uint32_t)(row * LDT + col) * 2u;
            asm volatile("global_load_async_to_lds_b128 %0, %1, off"
                         :: "v"(la), "v"(ga) : "memory");
            uint64_t gb = gBBase + kOff + ((uint64_t)row * KAUG + col) * 2u;
            uint32_t lb = bLds + (uint32_t)(row * LDT + col) * 2u;
            asm volatile("global_load_async_to_lds_b128 %0, %1, off"
                         :: "v"(lb), "v"(gb) : "memory");
        }
    };

    v8f acc0 = {};
    v8f acc1 = {};

    stage_load(0, 0);
    asm volatile("s_wait_asynccnt 0" ::: "memory");
    __syncthreads();

    for (int s = 0; s < nStages; ++s) {
        const int buf = s & 1;
        if (s + 1 < nStages) stage_load(buf ^ 1, s + 1);

        const _Float16* At = &lds[buf * 2 * TILE];
        const _Float16* Bl = &lds[buf * 2 * TILE + TILE];

#pragma unroll
        for (int kk = 0; kk < BK; kk += 32) {
            // A fragment (16x32 f16): lane = M row, 16 consecutive K per lane (laneHi*16)
            Frag fa;
            {
                const uint64_t* p0 =
                    (const uint64_t*)(At + (wm * 16 + laneLo) * LDT + kk + laneHi * 16);
                const uint64_t* p1 =
                    (const uint64_t*)(At + (wm * 16 + laneLo) * LDT + kk + laneHi * 16 + 8);
                fa.q[0] = p0[0]; fa.q[1] = p0[1];
                fa.q[2] = p1[0]; fa.q[3] = p1[1];
            }
            // B fragments (32x16 f16): lane = N column, 16 consecutive K per lane
            Frag fb0, fb1;
            {
                const uint64_t* p0 =
                    (const uint64_t*)(Bl + (wn * 32 + laneLo) * LDT + kk + laneHi * 16);
                fb0.q[0] = p0[0]; fb0.q[1] = p0[1]; fb0.q[2] = p0[2]; fb0.q[3] = p0[3];
                const uint64_t* p1 =
                    (const uint64_t*)(Bl + (wn * 32 + 16 + laneLo) * LDT + kk + laneHi * 16);
                fb1.q[0] = p1[0]; fb1.q[1] = p1[1]; fb1.q[2] = p1[2]; fb1.q[3] = p1[3];
            }
            acc0 = __builtin_amdgcn_wmma_f32_16x16x32_f16(
                false, fa.v, false, fb0.v, (short)0, acc0, false, false);
            acc1 = __builtin_amdgcn_wmma_f32_16x16x32_f16(
                false, fa.v, false, fb1.v, (short)0, acc1, false, false);
        }
        asm volatile("s_wait_asynccnt 0" ::: "memory");
        __syncthreads();
    }

    // C/D layout: VGPR r, lane -> M = r + laneHi*8, N = laneLo
    const int row0 = m0 + wm * 16 + laneHi * 8;
    const int col0 = n0 + wn * 32 + laneLo;
#pragma unroll
    for (int r = 0; r < 8; ++r) {
        float* p = out + (size_t)(row0 + r) * OUT_DIM + col0;
        unsafeAtomicAdd(p,      acc0[r]);
        unsafeAtomicAdd(p + 16, acc1[r]);
    }
}

// ---------------------------- launcher -------------------------------------

extern "C" void kernel_launch(void* const* d_in, const int* in_sizes, int n_in,
                              void* d_out, int out_size, void* d_ws, size_t ws_size,
                              hipStream_t stream) {
    const float* x    = (const float*)d_in[0];
    const float* grid = (const float*)d_in[1];
    const float* coef = (const float*)d_in[2];
    const float* sb   = (const float*)d_in[3];
    const float* ss   = (const float*)d_in[4];
    const float* mask = (const float*)d_in[5];
    float* out = (float*)d_out;

    _Float16* Aaug = (_Float16*)d_ws;
    _Float16* Bt   = (_Float16*)((char*)d_ws + (size_t)BATCH * KAUG * sizeof(_Float16));

    kan_zero_out<<<(BATCH * OUT_DIM) / 256, 256, 0, stream>>>(out);
    kan_prep_a<<<(BATCH * IN_DIM) / 256, 256, 0, stream>>>(x, grid, Aaug);
    kan_prep_b<<<(IN_DIM * OUT_DIM) / 256, 256, 0, stream>>>(coef, sb, ss, mask, Bt);
    kan_wmma_gemm<<<dim3(BATCH / BM, OUT_DIM / BN, SPLITK), 256, 0, stream>>>(Aaug, Bt, out);
}
